// MLPBlock_21354577395900
// MI455X (gfx1250) — matrix operand order repaired and checked by reading, not compiled, not run on hardware
//
#include <hip/hip_runtime.h>
#include <hip/hip_bf16.h>

#define T_TOK 2048
#define HDIM  2048
#define IDIM  2048
#define NEXP  8

typedef __bf16 bf16_t;
typedef __attribute__((ext_vector_type(16))) __bf16 v16bf;
typedef __attribute__((ext_vector_type(8)))  float  v8f;

// ---- fragment loaders (contiguous k = half*16 + j, same order for A and B) ----

// A 16x32 bf16: lane half=0 -> K {0..15}; half=1 -> K {16..31}
__device__ __forceinline__ v16bf load_a_frag(const bf16_t* row, int k0, int half) {
    union { uint4 u[2]; v16bf v; } t;
    t.u[0] = *(const uint4*)(row + k0 + half * 16);
    t.u[1] = *(const uint4*)(row + k0 + half * 16 + 8);
    return t.v;
}

// B 32x16 bf16 from fp32 weights: lane col c = lane&15, K = half*16 + j (j=0..15 contiguous)
__device__ __forceinline__ v16bf load_b_frag_f32(const float* wrow_k) {
    union { v16bf v; bf16_t h[16]; } t;
    const float4* p = (const float4*)wrow_k;
#pragma unroll
    for (int j = 0; j < 4; ++j) {
        float4 f = p[j];
        t.h[4 * j + 0] = (bf16_t)f.x;
        t.h[4 * j + 1] = (bf16_t)f.y;
        t.h[4 * j + 2] = (bf16_t)f.z;
        t.h[4 * j + 3] = (bf16_t)f.w;
    }
    return t.v;
}

__device__ __forceinline__ v8f wmma_bf16(v16bf a, v16bf b, v8f c) {
    return __builtin_amdgcn_wmma_f32_16x16x32_bf16(false, a, false, b, (short)0, c, false, false);
}

// ---- kernel 1: fp32 -> bf16 copy of x ----
__global__ __launch_bounds__(256) void k_cvt_x(const float* __restrict__ x,
                                               bf16_t* __restrict__ xbf, int n4) {
    int i = blockIdx.x * blockDim.x + threadIdx.x;
    if (i >= n4) return;
    float4 f = ((const float4*)x)[i];
    union { bf16_t h[4]; uint2 u; } t;
    t.h[0] = (bf16_t)f.x; t.h[1] = (bf16_t)f.y; t.h[2] = (bf16_t)f.z; t.h[3] = (bf16_t)f.w;
    ((uint2*)xbf)[i] = t.u;
}

// ---- kernel 2: router. one wave per token: 8 dots of length H, softmax top-2 ----
__global__ __launch_bounds__(32) void k_router(const float* __restrict__ x,
                                               const float* __restrict__ rw,
                                               const float* __restrict__ rb,
                                               int* __restrict__ topE,
                                               float* __restrict__ wslot) {
    int t = blockIdx.x;
    int lane = threadIdx.x;
    const float* xr = x + (size_t)t * HDIM;
    float acc[NEXP];
#pragma unroll
    for (int e = 0; e < NEXP; ++e) acc[e] = 0.f;
    for (int h = lane; h < HDIM; h += 32) {
        float xv = xr[h];
#pragma unroll
        for (int e = 0; e < NEXP; ++e) acc[e] += xv * rw[e * HDIM + h];
    }
#pragma unroll
    for (int e = 0; e < NEXP; ++e) {
#pragma unroll
        for (int off = 16; off > 0; off >>= 1) acc[e] += __shfl_xor(acc[e], off, 32);
    }
    if (lane == 0) {
        float lg[NEXP];
#pragma unroll
        for (int e = 0; e < NEXP; ++e) lg[e] = acc[e] + rb[e];
        int e0 = 0;
#pragma unroll
        for (int e = 1; e < NEXP; ++e) if (lg[e] > lg[e0]) e0 = e;
        int e1 = -1;
#pragma unroll
        for (int e = 0; e < NEXP; ++e) {
            if (e == e0) continue;
            if (e1 < 0 || lg[e] > lg[e1]) e1 = e;
        }
        // renormalized top-2 softmax weights: w0 = 1/(1+exp(l1-l0))
        float w0 = 1.f / (1.f + __expf(lg[e1] - lg[e0]));
        topE[2 * t] = e0;  topE[2 * t + 1] = e1;
        wslot[2 * t] = w0; wslot[2 * t + 1] = 1.f - w0;
    }
}

// ---- kernel 3: deterministic per-expert token lists via wave32 ballot compaction ----
__global__ __launch_bounds__(32) void k_build_lists(const int* __restrict__ topE,
                                                    int* __restrict__ list,
                                                    int* __restrict__ counts) {
    int e = blockIdx.x;
    int lane = threadIdx.x;
    int total = 0;
    for (int base = 0; base < T_TOK; base += 32) {
        int t = base + lane;
        bool m1 = (topE[2 * t + 1] == e);
        bool m = (topE[2 * t] == e) || m1;
        unsigned mask = (unsigned)__ballot(m);
        if (m) {
            int prefix = __popc(mask & ((1u << lane) - 1u));
            list[e * T_TOK + total + prefix] = (t << 1) | (m1 ? 1 : 0);
        }
        total += __popc(mask);
    }
    if (lane == 0) counts[e] = total;
}

// ---- kernel 4: grouped GEMM1 + SiLU-gate.
// block: 8 waves, tile M=32 tokens (two 16-row subtiles sharing B fragments),
// each wave: 32 gate cols + 32 matching up cols -> 8 accumulators
__global__ __launch_bounds__(256) void k_gemm1(const bf16_t* __restrict__ xbf,
                                               const float* __restrict__ w13,
                                               const float* __restrict__ b13,
                                               const int* __restrict__ list,
                                               const int* __restrict__ counts,
                                               bf16_t* __restrict__ hbuf) {
    int e = blockIdx.z;
    int cnt = counts[e];
    int row_base = blockIdx.y * 32;
    if (row_base >= cnt) return;

    __shared__ int s_entry[32];
    if (threadIdx.x < 32) {
        int idx = row_base + (int)threadIdx.x;
        if (idx >= cnt) idx = cnt - 1;
        s_entry[threadIdx.x] = list[e * T_TOK + idx];
    }
    __syncthreads();

    int wave = threadIdx.x >> 5;
    int lane = threadIdx.x & 31;
    int half = lane >> 4;
    int c = lane & 15;
    int jb = blockIdx.x * 256 + wave * 32;      // feature base within [0, IDIM)

    const bf16_t* xrow0 = xbf + (size_t)(s_entry[c] >> 1) * HDIM;
    const bf16_t* xrow1 = xbf + (size_t)(s_entry[16 + c] >> 1) * HDIM;
    const float* wg0 = w13 + ((size_t)e * (2 * IDIM) + (jb + c)) * HDIM;
    const float* wg1 = wg0 + (size_t)16 * HDIM;
    const float* wu0 = w13 + ((size_t)e * (2 * IDIM) + IDIM + (jb + c)) * HDIM;
    const float* wu1 = wu0 + (size_t)16 * HDIM;

    v8f g0a = {}, g1a = {}, u0a = {}, u1a = {};
    v8f g0b = {}, g1b = {}, u0b = {}, u1b = {};
    for (int k0 = 0; k0 < HDIM; k0 += 32) {
        v16bf a0 = load_a_frag(xrow0, k0, half);
        v16bf a1 = load_a_frag(xrow1, k0, half);
        int ko = k0 + half * 16;
        v16bf b;
        b = load_b_frag_f32(wg0 + ko); g0a = wmma_bf16(a0, b, g0a); g0b = wmma_bf16(a1, b, g0b);
        b = load_b_frag_f32(wg1 + ko); g1a = wmma_bf16(a0, b, g1a); g1b = wmma_bf16(a1, b, g1b);
        b = load_b_frag_f32(wu0 + ko); u0a = wmma_bf16(a0, b, u0a); u0b = wmma_bf16(a1, b, u0b);
        b = load_b_frag_f32(wu1 + ko); u1a = wmma_bf16(a0, b, u1a); u1b = wmma_bf16(a1, b, u1b);
    }

    float bg0 = b13[e * (2 * IDIM) + jb + c];
    float bg1 = b13[e * (2 * IDIM) + jb + 16 + c];
    float bu0 = b13[e * (2 * IDIM) + IDIM + jb + c];
    float bu1 = b13[e * (2 * IDIM) + IDIM + jb + 16 + c];

#pragma unroll
    for (int tile = 0; tile < 2; ++tile) {
        const v8f& G0 = tile ? g0b : g0a;
        const v8f& G1 = tile ? g1b : g1a;
        const v8f& U0 = tile ? u0b : u0a;
        const v8f& U1 = tile ? u1b : u1a;
#pragma unroll
        for (int v = 0; v < 8; ++v) {
            int m = tile * 16 + half * 8 + v;   // C/D layout: row m = (lane>>4)*8 + vgpr
            if (row_base + m < cnt) {
                bf16_t* hrow = hbuf + (size_t)s_entry[m] * IDIM;
                float g = G0[v] + bg0, u = U0[v] + bu0;
                hrow[jb + c] = (bf16_t)((g / (1.f + __expf(-g))) * u);
                g = G1[v] + bg1; u = U1[v] + bu1;
                hrow[jb + 16 + c] = (bf16_t)((g / (1.f + __expf(-g))) * u);
            }
        }
    }
}

// ---- kernel 5: grouped GEMM2, bias + routing weight, scatter to per-slot y buffer.
// block: 8 waves, tile M=32 (two 16-row subtiles), each wave 64 output cols -> 8 accumulators
__global__ __launch_bounds__(256) void k_gemm2(const bf16_t* __restrict__ hbuf,
                                               const float* __restrict__ w2,
                                               const float* __restrict__ b2,
                                               const int* __restrict__ list,
                                               const int* __restrict__ counts,
                                               const float* __restrict__ wslot,
                                               float* __restrict__ yslot) {
    int e = blockIdx.z;
    int cnt = counts[e];
    int row_base = blockIdx.y * 32;
    if (row_base >= cnt) return;

    __shared__ int s_entry[32];
    __shared__ float s_w[32];
    if (threadIdx.x < 32) {
        int idx = row_base + (int)threadIdx.x;
        if (idx >= cnt) idx = cnt - 1;
        int en = list[e * T_TOK + idx];
        s_entry[threadIdx.x] = en;
        s_w[threadIdx.x] = wslot[en];
    }
    __syncthreads();

    int wave = threadIdx.x >> 5;
    int lane = threadIdx.x & 31;
    int half = lane >> 4;
    int c = lane & 15;
    int hb = blockIdx.x * 512 + wave * 64;      // output col base within [0, HDIM)

    const bf16_t* arow0 = hbuf + (size_t)s_entry[c] * IDIM;
    const bf16_t* arow1 = hbuf + (size_t)s_entry[16 + c] * IDIM;
    const float* wr = w2 + ((size_t)e * HDIM + hb + c) * IDIM;

    v8f acc0[4] = {{}, {}, {}, {}};
    v8f acc1[4] = {{}, {}, {}, {}};
    for (int k0 = 0; k0 < IDIM; k0 += 32) {
        v16bf a0 = load_a_frag(arow0, k0, half);
        v16bf a1 = load_a_frag(arow1, k0, half);
        int ko = k0 + half * 16;
#pragma unroll
        for (int f = 0; f < 4; ++f) {
            v16bf b = load_b_frag_f32(wr + (size_t)f * 16 * IDIM + ko);
            acc0[f] = wmma_bf16(a0, b, acc0[f]);
            acc1[f] = wmma_bf16(a1, b, acc1[f]);
        }
    }

#pragma unroll
    for (int f = 0; f < 4; ++f) {
        float bias = b2[e * HDIM + hb + f * 16 + c];
#pragma unroll
        for (int v = 0; v < 8; ++v) {
            int m0 = half * 8 + v;
            if (row_base + m0 < cnt)
                yslot[(size_t)s_entry[m0] * HDIM + hb + f * 16 + c] = (acc0[f][v] + bias) * s_w[m0];
            int m1 = 16 + half * 8 + v;
            if (row_base + m1 < cnt)
                yslot[(size_t)s_entry[m1] * HDIM + hb + f * 16 + c] = (acc1[f][v] + bias) * s_w[m1];
        }
    }
}

// ---- kernel 6: deterministic combine (no atomics): out[t] = y[t,slot0] + y[t,slot1] ----
__global__ __launch_bounds__(256) void k_combine(const float* __restrict__ yslot,
                                                 float* __restrict__ out, int n4) {
    int i = blockIdx.x * blockDim.x + threadIdx.x;
    if (i >= n4) return;
    int idx = i * 4;
    int t = idx / HDIM;
    int h = idx - t * HDIM;
    float4 a = *(const float4*)(yslot + ((size_t)(2 * t)) * HDIM + h);
    float4 b = *(const float4*)(yslot + ((size_t)(2 * t) + 1) * HDIM + h);
    float4 r; r.x = a.x + b.x; r.y = a.y + b.y; r.z = a.z + b.z; r.w = a.w + b.w;
    ((float4*)out)[i] = r;
}

extern "C" void kernel_launch(void* const* d_in, const int* in_sizes, int n_in,
                              void* d_out, int out_size, void* d_ws, size_t ws_size,
                              hipStream_t stream) {
    const float* x        = (const float*)d_in[0];
    const float* router_w = (const float*)d_in[1];
    const float* router_b = (const float*)d_in[2];
    const float* w13      = (const float*)d_in[3];
    const float* b13      = (const float*)d_in[4];
    const float* w2       = (const float*)d_in[5];
    const float* b2       = (const float*)d_in[6];
    float* out = (float*)d_out;

    // workspace carve-up (256B aligned): ~56 MB total
    char* p = (char*)d_ws;
    auto carve = [&](size_t bytes) -> void* {
        void* r = (void*)p;
        p += (bytes + 255) & ~(size_t)255;
        return r;
    };
    int*    counts = (int*)   carve(NEXP * sizeof(int));
    int*    topE   = (int*)   carve((size_t)T_TOK * 2 * sizeof(int));
    float*  wslot  = (float*) carve((size_t)T_TOK * 2 * sizeof(float));
    int*    list   = (int*)   carve((size_t)NEXP * T_TOK * sizeof(int));
    bf16_t* xbf    = (bf16_t*)carve((size_t)T_TOK * HDIM * sizeof(bf16_t));
    bf16_t* hbuf   = (bf16_t*)carve((size_t)2 * T_TOK * IDIM * sizeof(bf16_t));
    float*  yslot  = (float*) carve((size_t)2 * T_TOK * HDIM * sizeof(float));

    int n4x = T_TOK * HDIM / 4;
    k_cvt_x<<<(n4x + 255) / 256, 256, 0, stream>>>(x, xbf, n4x);
    k_router<<<T_TOK, 32, 0, stream>>>(x, router_w, router_b, topE, wslot);
    k_build_lists<<<NEXP, 32, 0, stream>>>(topE, list, counts);

    dim3 g1(IDIM / 256, T_TOK / 32, NEXP);   // expert in z (slowest) for L2 weight reuse
    k_gemm1<<<g1, 256, 0, stream>>>(xbf, w13, b13, list, counts, hbuf);

    dim3 g2(HDIM / 512, T_TOK / 32, NEXP);
    k_gemm2<<<g2, 256, 0, stream>>>(hbuf, w2, b2, list, counts, wslot, yslot);

    k_combine<<<(n4x + 255) / 256, 256, 0, stream>>>(yslot, out, n4x);
}
